// GATModel_18373870092585
// MI455X (gfx1250) — matrix operand order repaired and checked by reading, not compiled, not run on hardware
//
#include <hip/hip_runtime.h>
#include <hip/hip_bf16.h>

typedef __bf16 bf16;
typedef __attribute__((ext_vector_type(8)))  __bf16 bf16x8;
typedef __attribute__((ext_vector_type(16))) __bf16 bf16x16;
typedef __attribute__((ext_vector_type(8)))  float  floatx8;

#define HEADS 4
#define CH    64
#define HC    256

// ---------------- conversion / packing ----------------

__global__ void k_f32_to_bf16(const float* __restrict__ in, bf16* __restrict__ out, int n) {
  int i = blockIdx.x * blockDim.x + threadIdx.x;
  if (i < n) out[i] = (bf16)in[i];
}

// Pack W[K,Nn] (row-major f32) into WMMA B-fragment order (bf16):
// Bp[((kt*tilesN + tn)*32 + lane)*16 + i] = W[kt*32 + (lane>>4)*16 + i][tn*16 + (lane&15)]
__global__ void k_pack_w(const float* __restrict__ W, bf16* __restrict__ Bp, int K, int Nn) {
  int idx = blockIdx.x * blockDim.x + threadIdx.x;
  int tilesN = Nn >> 4;
  int total = (K >> 5) * tilesN * 32 * 16;
  if (idx >= total) return;
  int i    = idx & 15;
  int lane = (idx >> 4) & 31;
  int t    = idx >> 9;
  int tn   = t % tilesN;
  int kt   = t / tilesN;
  int k = kt * 32 + (lane >> 4) * 16 + i;
  int n = tn * 16 + (lane & 15);
  Bp[idx] = (bf16)W[(size_t)k * Nn + n];
}

// ---------------- WMMA GEMM: C[M,Nn] = A[M,K] @ W (Bp packed) ----------------
// Each wave computes NT adjacent 16x16 N-tiles, reusing one A fragment per
// K-step across NT v_wmma issues (4x arithmetic intensity on the A stream).

template <int NT>
__global__ void __launch_bounds__(128)
k_wmma_gemm(const bf16* __restrict__ A, const bf16* __restrict__ Bp,
            float* __restrict__ C, int M, int Nn, int K) {
  const int tilesN  = Nn >> 4;
  const int groupsN = tilesN / NT;
  const int tilesM  = (M + 15) >> 4;
  const int wave = threadIdx.x >> 5;
  const int lane = threadIdx.x & 31;
  const int g = blockIdx.x * 4 + wave;
  if (g >= tilesM * groupsN) return;           // wave-uniform: EXEC stays full
  const int tm = g / groupsN;
  const int tg = g % groupsN;                  // N-tile group [tg*NT, tg*NT+NT)
  const int half = lane >> 4;
  const int l15  = lane & 15;
  int row = tm * 16 + l15;
  int rowc = row < M ? row : M - 1;            // clamp keeps EXEC all-ones

  floatx8 acc[NT];
  #pragma unroll
  for (int j = 0; j < NT; ++j) acc[j] = (floatx8){};

  const int ksteps = K >> 5;
  for (int ks = 0; ks < ksteps; ++ks) {
    const int k0 = ks << 5;
    // A fragment: lane holds row l15, K = k0 + half*16 + {0..15} (matches packed B)
    const bf16* arow = A + (size_t)rowc * K + k0 + half * 16;
    bf16x8 alo = *(const bf16x8*)(arow);
    bf16x8 ahi = *(const bf16x8*)(arow + 8);
    bf16x16 a = __builtin_shufflevector(alo, ahi, 0,1,2,3,4,5,6,7,8,9,10,11,12,13,14,15);
    #pragma unroll
    for (int j = 0; j < NT; ++j) {
      const bf16* brow = Bp + (((size_t)ks * tilesN + tg * NT + j) * 32 + lane) * 16;
      bf16x8 blo = *(const bf16x8*)(brow);
      bf16x8 bhi = *(const bf16x8*)(brow + 8);
      bf16x16 b = __builtin_shufflevector(blo, bhi, 0,1,2,3,4,5,6,7,8,9,10,11,12,13,14,15);
      acc[j] = __builtin_amdgcn_wmma_f32_16x16x32_bf16(false, a, false, b, (short)0,
                                                       acc[j], false, false);
    }
  }
  // C/D layout: VGPR r -> row tm*16 + half*8 + r, col tile*16 + l15
  #pragma unroll
  for (int j = 0; j < NT; ++j) {
    int col = (tg * NT + j) * 16 + l15;
    #pragma unroll
    for (int r = 0; r < 8; ++r) {
      int orow = tm * 16 + half * 8 + r;
      if (orow < M) C[(size_t)orow * Nn + col] = acc[j][r];
    }
  }
}

// ---------------- attention score dots: es/ed [N,H] ----------------

__global__ void k_scores(const float* __restrict__ h, const float* __restrict__ asrc,
                         const float* __restrict__ adst, float* __restrict__ es,
                         float* __restrict__ ed, int NH) {
  int idx = blockIdx.x * blockDim.x + threadIdx.x;
  if (idx >= NH) return;
  int n  = idx >> 2;
  int hh = idx & 3;
  const float* hp = h + (size_t)n * HC + hh * CH;
  float s1 = 0.f, s2 = 0.f;
  #pragma unroll 8
  for (int c = 0; c < CH; ++c) {
    float v = hp[c];
    s1 += v * asrc[hh * CH + c];
    s2 += v * adst[hh * CH + c];
  }
  es[idx] = s1;
  ed[idx] = s2;
}

// ---------------- per-layer init ----------------

__global__ void k_init_layer(float* __restrict__ agg, float* __restrict__ m,
                             float* __restrict__ denom, int N) {
  int idx = blockIdx.x * blockDim.x + threadIdx.x;
  if (idx < N * HC) agg[idx] = 0.f;
  if (idx < N * HEADS) {
    m[idx] = -__builtin_inff();
    denom[idx] = 0.f;
  }
}

// ---------------- edge kernels ----------------

__device__ __forceinline__ void atomicMaxF(float* addr, float v) {
  if (v >= 0.f) atomicMax((int*)addr, __float_as_int(v));
  else          atomicMin((unsigned int*)addr, __float_as_uint(v));
}

__global__ void k_edge_logits(const int* __restrict__ src, const int* __restrict__ dst,
                              const float* __restrict__ es, const float* __restrict__ ed,
                              float* __restrict__ ebuf, float* __restrict__ m,
                              int E, int Etot) {
  int idx = blockIdx.x * blockDim.x + threadIdx.x;
  if (idx >= Etot * HEADS) return;
  int e  = idx >> 2;
  int hh = idx & 3;
  int s, d;
  if (e < E) { s = src[e]; d = dst[e]; } else { s = d = e - E; }
  float v = es[s * HEADS + hh] + ed[d * HEADS + hh];
  v = v > 0.f ? v : 0.2f * v;          // leaky_relu(0.2)
  ebuf[idx] = v;
  atomicMaxF(&m[d * HEADS + hh], v);
}

__global__ void k_edge_exp(const int* __restrict__ dst, float* __restrict__ ebuf,
                           const float* __restrict__ m, float* __restrict__ denom,
                           int E, int Etot) {
  int idx = blockIdx.x * blockDim.x + threadIdx.x;
  if (idx >= Etot * HEADS) return;
  int e  = idx >> 2;
  int hh = idx & 3;
  int d = (e < E) ? dst[e] : e - E;
  float mm = m[d * HEADS + hh];
  if (!(mm > -3.0e38f && mm < 3.0e38f)) mm = 0.f;  // isfinite guard
  float ex = __expf(ebuf[idx] - mm);
  ebuf[idx] = ex;
  atomicAdd(&denom[d * HEADS + hh], ex);
}

// 4 edges per 256-thread block; 64 threads/edge, one float4 (b128) gather each.
__global__ void __launch_bounds__(256)
k_edge_aggr(const int* __restrict__ src, const int* __restrict__ dst,
            const float* __restrict__ ebuf, const float* __restrict__ denom,
            const float* __restrict__ h, float* __restrict__ agg, int E, int Etot) {
  int e = blockIdx.x * 4 + (threadIdx.x >> 6);
  if (e >= Etot) return;
  int t = threadIdx.x & 63;            // 64 threads cover HC=256 via float4
  int s, d;
  if (e < E) { s = src[e]; d = dst[e]; } else { s = d = e - E; }
  int c  = t << 2;                     // channel base
  int hh = c >> 6;                     // head index
  float alpha = ebuf[e * HEADS + hh] / fmaxf(denom[d * HEADS + hh], 1e-16f);
  const float4 hv = *(const float4*)(h + (size_t)s * HC + c);
  float* ap = agg + (size_t)d * HC + c;
  atomicAdd(ap + 0, alpha * hv.x);
  atomicAdd(ap + 1, alpha * hv.y);
  atomicAdd(ap + 2, alpha * hv.z);
  atomicAdd(ap + 3, alpha * hv.w);
}

// ---------------- bias + relu -> bf16 for next GEMM ----------------

__global__ void k_bias_relu_bf16(const float* __restrict__ agg, const float* __restrict__ b,
                                 bf16* __restrict__ xb, int tot) {
  int idx = blockIdx.x * blockDim.x + threadIdx.x;
  if (idx >= tot) return;
  float v = agg[idx] + b[idx & (HC - 1)];
  xb[idx] = (bf16)fmaxf(v, 0.f);
}

// ---------------- pooling ----------------

__global__ void k_zero_pool(float* __restrict__ sums, float* __restrict__ counts, int G) {
  int idx = blockIdx.x * blockDim.x + threadIdx.x;
  if (idx < G * 32) sums[idx] = 0.f;
  if (idx < G) counts[idx] = 0.f;
}

__global__ void k_pool(const int* __restrict__ batch, const float* __restrict__ scores,
                       float* __restrict__ sums, float* __restrict__ counts, int N) {
  int idx = blockIdx.x * blockDim.x + threadIdx.x;
  if (idx >= N * 32) return;
  int n  = idx >> 5;
  int dd = idx & 31;
  int g = batch[n];
  atomicAdd(&sums[g * 32 + dd], scores[idx]);
  if (dd == 0) atomicAdd(&counts[g], 1.0f);
}

__global__ void k_finalize(const float* __restrict__ sums, const float* __restrict__ counts,
                           const float* __restrict__ bp, float* __restrict__ out, int G) {
  int idx = blockIdx.x * blockDim.x + threadIdx.x;
  if (idx >= G * 32) return;
  int g  = idx >> 5;
  int dd = idx & 31;
  float cnt = counts[g];
  out[idx] = (sums[idx] + cnt * bp[dd]) / fmaxf(cnt, 1.0f);
}

// ---------------- host orchestration ----------------

static inline int cdiv(int a, int b) { return (a + b - 1) / b; }

extern "C" void kernel_launch(void* const* d_in, const int* in_sizes, int n_in,
                              void* d_out, int out_size, void* d_ws, size_t ws_size,
                              hipStream_t stream) {
  const float* x      = (const float*)d_in[0];
  const int*   eidx   = (const int*)d_in[1];
  const int*   batch  = (const int*)d_in[2];
  const float* W1     = (const float*)d_in[3];
  const float* a_src1 = (const float*)d_in[4];
  const float* a_dst1 = (const float*)d_in[5];
  const float* b1     = (const float*)d_in[6];
  const float* W2     = (const float*)d_in[7];
  const float* a_src2 = (const float*)d_in[8];
  const float* a_dst2 = (const float*)d_in[9];
  const float* b2     = (const float*)d_in[10];
  const float* Wp     = (const float*)d_in[11];
  const float* bp     = (const float*)d_in[12];
  float* out = (float*)d_out;

  const int N    = in_sizes[0] / HC;   // D_IN == HC == 256
  const int E    = in_sizes[1] / 2;
  const int Etot = E + N;              // with self loops
  const int G    = out_size / 32;

  // workspace carve-out
  char* ws = (char*)d_ws;
  auto alloc = [&](size_t bytes) -> char* {
    char* p = ws;
    ws += (bytes + 255) & ~(size_t)255;
    return p;
  };
  bf16*  x_bf  = (bf16*)alloc((size_t)N * HC * sizeof(bf16));   // reused per layer
  bf16*  W1p   = (bf16*)alloc((size_t)HC * HC * sizeof(bf16));
  bf16*  W2p   = (bf16*)alloc((size_t)HC * HC * sizeof(bf16));
  bf16*  Wpp   = (bf16*)alloc((size_t)HC * 32 * sizeof(bf16));
  float* h     = (float*)alloc((size_t)N * HC * sizeof(float)); // GEMM out (both layers)
  float* agg   = (float*)alloc((size_t)N * HC * sizeof(float));
  float* es    = (float*)alloc((size_t)N * HEADS * sizeof(float));
  float* ed    = (float*)alloc((size_t)N * HEADS * sizeof(float));
  float* m     = (float*)alloc((size_t)N * HEADS * sizeof(float));
  float* denom = (float*)alloc((size_t)N * HEADS * sizeof(float));
  float* ebuf  = (float*)alloc((size_t)Etot * HEADS * sizeof(float));
  float* score = (float*)alloc((size_t)N * 32 * sizeof(float));
  float* sums  = (float*)alloc((size_t)G * 32 * sizeof(float));
  float* cnts  = (float*)alloc((size_t)G * sizeof(float));

  const int* srcI = eidx;
  const int* dstI = eidx + E;

  const int T = 256;
  // convert / pack weights
  k_f32_to_bf16<<<cdiv(N * HC, T), T, 0, stream>>>(x, x_bf, N * HC);
  k_pack_w<<<cdiv(HC * HC, T), T, 0, stream>>>(W1, W1p, HC, HC);
  k_pack_w<<<cdiv(HC * HC, T), T, 0, stream>>>(W2, W2p, HC, HC);
  k_pack_w<<<cdiv(HC * 32, T), T, 0, stream>>>(Wp, Wpp, HC, 32);

  const int tilesM = (N + 15) / 16;
  const int gemmBlocksHC = cdiv(tilesM * (HC / 16 / 4), 4);  // NT=4 groups
  const int gemmBlocks32 = cdiv(tilesM * (32 / 16 / 2), 4);  // NT=2 groups
  const int edgeHT = cdiv(Etot * HEADS, T);

  for (int layer = 0; layer < 2; ++layer) {
    const float* asrc = layer == 0 ? a_src1 : a_src2;
    const float* adst = layer == 0 ? a_dst1 : a_dst2;
    const float* bb   = layer == 0 ? b1 : b2;
    const bf16*  Wpk  = layer == 0 ? W1p : W2p;

    k_init_layer<<<cdiv(N * HC, T), T, 0, stream>>>(agg, m, denom, N);
    k_wmma_gemm<4><<<gemmBlocksHC, 128, 0, stream>>>(x_bf, Wpk, h, N, HC, HC);
    k_scores<<<cdiv(N * HEADS, T), T, 0, stream>>>(h, asrc, adst, es, ed, N * HEADS);
    k_edge_logits<<<edgeHT, T, 0, stream>>>(srcI, dstI, es, ed, ebuf, m, E, Etot);
    k_edge_exp<<<edgeHT, T, 0, stream>>>(dstI, ebuf, m, denom, E, Etot);
    k_edge_aggr<<<cdiv(Etot, 4), 256, 0, stream>>>(srcI, dstI, ebuf, denom, h, agg, E, Etot);
    k_bias_relu_bf16<<<cdiv(N * HC, T), T, 0, stream>>>(agg, bb, x_bf, N * HC);
  }

  // projection GEMM: scores = x2 @ Wp
  k_wmma_gemm<2><<<gemmBlocks32, 128, 0, stream>>>(x_bf, Wpp, score, N, 32, HC);

  // mean pool per graph (+bp folded in analytically)
  k_zero_pool<<<cdiv(G * 32, T), T, 0, stream>>>(sums, cnts, G);
  k_pool<<<cdiv(N * 32, T), T, 0, stream>>>(batch, score, sums, cnts, N);
  k_finalize<<<cdiv(G * 32, T), T, 0, stream>>>(sums, cnts, bp, out, G);
}
